// HindiCausalLMSelfAttention_24979529794093
// MI455X (gfx1250) — matrix operand order repaired and checked by reading, not compiled, not run on hardware
//
#include <hip/hip_runtime.h>
#include <hip/hip_bf16.h>

typedef __attribute__((ext_vector_type(16))) __bf16 v16bf;
typedef __attribute__((ext_vector_type(8)))  float  v8f;
typedef unsigned short u16;
typedef unsigned int   u32;
typedef unsigned long long u64;

typedef __attribute__((ext_vector_type(4))) unsigned int u32x4;
typedef __attribute__((ext_vector_type(8))) int          i32x8;
typedef __attribute__((ext_vector_type(4))) int          i32x4;

union FragBF { v16bf v; uint4 q[2]; u16 h[16]; };
union FragF  { v8f   v; float f[8]; };
union Pack8  { uint4 q; u16 h[8]; };

__device__ __forceinline__ u16 f2bf(float x) {
  u32 u = __builtin_bit_cast(u32, x);
  u32 lsb = (u >> 16) & 1u;
  u += 0x7fffu + lsb;                 // round-to-nearest-even
  return (u16)(u >> 16);
}
__device__ __forceinline__ float bf2f(u16 h) {
  return __builtin_bit_cast(float, (u32)h << 16);
}

#define WMMA_BF16(A, B, C) \
  __builtin_amdgcn_wmma_f32_16x16x32_bf16(false, (A), false, (B), (short)0, (C), false, false)

// --------------------------- TDM support -----------------------------------
#if __has_builtin(__builtin_amdgcn_tensor_load_to_lds)
#define HAVE_TDM 1
#else
#define HAVE_TDM 0
#endif

#if HAVE_TDM
// 2D tensor tile -> LDS, data_size = 2 bytes, LDS padding enabled.
// pad_iv: pad after 2^(pad_iv+1) DWORDs; pad_am: pad (pad_am+1) DWORDs.
__device__ __forceinline__ void tdm_load_2d(u32 lds_off, const void* gptr,
                                            u32 td0, u32 td1, u32 tile0,
                                            u32 tile1, u32 stride0,
                                            u32 pad_iv, u32 pad_am)
{
  const u64 ga = (u64)(size_t)gptr;
  u32x4 g0;
  g0[0] = 1u;                                   // count=1, user descriptor
  g0[1] = lds_off;                              // LDS byte address
  g0[2] = (u32)ga;                              // global_addr[31:0]
  g0[3] = (u32)((ga >> 32) & 0x01FFFFFFu) | (2u << 30);   // addr[56:32], type=2
  i32x8 g1;
  g1[0] = (int)((1u << 16) | (1u << 20) | (pad_iv << 22) | (pad_am << 25));
  g1[1] = (int)((td0 & 0xFFFFu) << 16);         // tensor_dim0[15:0]
  g1[2] = (int)((td0 >> 16) | ((td1 & 0xFFFFu) << 16));
  g1[3] = (int)((td1 >> 16) | (tile0 << 16));   // tile_dim0
  g1[4] = (int)(tile1 & 0xFFFFu);               // tile_dim1 (tile_dim2=0)
  g1[5] = (int)stride0;                         // tensor_dim0_stride[31:0]
  g1[6] = 0;
  g1[7] = 0;
  i32x4 z4 = {0, 0, 0, 0};
#if __clang_major__ >= 23
  i32x8 z8 = {0, 0, 0, 0, 0, 0, 0, 0};
  __builtin_amdgcn_tensor_load_to_lds(g0, g1, z4, z4, z8, 0);
#else
  __builtin_amdgcn_tensor_load_to_lds(g0, g1, z4, z4, 0);
#endif
}

__device__ __forceinline__ void tdm_wait0() {
#if __has_builtin(__builtin_amdgcn_s_wait_tensorcnt)
  __builtin_amdgcn_s_wait_tensorcnt(0);
#else
  asm volatile("s_wait_tensorcnt 0x0" ::: "memory");
#endif
}
#endif  // HAVE_TDM

// ---------------------------------------------------------------------------
// GEMM: out[M,N] = A[M,K](f32) * W[K,N](f32) + bias[N], bf16 math, f32 accum.
// 64x128 tile per 256-thread block (8 waves, each wave = 16x64 of output:
// one A fragment reused across 4 B fragments -> 4 WMMAs per K-step).
// ---------------------------------------------------------------------------
template<bool OUT_F32>
__global__ __launch_bounds__(256)
void gemm64_bias(const float* __restrict__ A, const float* __restrict__ W,
                 const float* __restrict__ bias, void* __restrict__ outp,
                 int M, int N, int K)
{
  __shared__ u16 Al[64][40];    // [m][k] rows padded to 80B
  __shared__ u16 Bl[128][40];   // [n][k]

  const int tid   = threadIdx.x;
  const int lane  = tid & 31;
  const int wid   = tid >> 5;
  const int laneN = lane & 15;
  const int hi    = lane >> 4;
  const int m0 = blockIdx.y * 64;
  const int n0 = blockIdx.x * 128;
  const int mw = (wid >> 1) * 16;     // wave row offset in tile
  const int nw = (wid & 1) * 64;      // wave col offset in tile

  FragF c[4];
#pragma unroll
  for (int t = 0; t < 4; ++t)
#pragma unroll
    for (int i = 0; i < 8; ++i) c[t].f[i] = 0.f;

  const int arow = tid >> 2, aseg = tid & 3;   // 64 rows x 4 segs of 8
  const int bkr  = tid >> 3, bns  = tid & 7;   // 32 k-rows x 8 n-segs of 16

  for (int k0 = 0; k0 < K; k0 += 32) {
    __syncthreads();
    { // A tile 64x32 f32 -> bf16
      const float* src = A + (size_t)(m0 + arow) * K + (k0 + aseg * 8);
      float4 f0 = *(const float4*)src;
      float4 f1 = *(const float4*)(src + 4);
      if (k0 + 32 < K) __builtin_prefetch(src + 32, 0, 1);
      Pack8 p;
      p.h[0]=f2bf(f0.x); p.h[1]=f2bf(f0.y); p.h[2]=f2bf(f0.z); p.h[3]=f2bf(f0.w);
      p.h[4]=f2bf(f1.x); p.h[5]=f2bf(f1.y); p.h[6]=f2bf(f1.z); p.h[7]=f2bf(f1.w);
      *(uint4*)&Al[arow][aseg * 8] = p.q;
    }
    { // W tile 32x128 f32 -> bf16, transposed into Bl[n][k]
      const float* src = W + (size_t)(k0 + bkr) * N + (n0 + bns * 16);
      float4 f0 = *(const float4*)src;
      float4 f1 = *(const float4*)(src + 4);
      float4 f2 = *(const float4*)(src + 8);
      float4 f3 = *(const float4*)(src + 12);
      if (k0 + 32 < K) __builtin_prefetch(src + (size_t)32 * N, 0, 1);
      const int nb = bns * 16;
      Bl[nb+ 0][bkr]=f2bf(f0.x); Bl[nb+ 1][bkr]=f2bf(f0.y);
      Bl[nb+ 2][bkr]=f2bf(f0.z); Bl[nb+ 3][bkr]=f2bf(f0.w);
      Bl[nb+ 4][bkr]=f2bf(f1.x); Bl[nb+ 5][bkr]=f2bf(f1.y);
      Bl[nb+ 6][bkr]=f2bf(f1.z); Bl[nb+ 7][bkr]=f2bf(f1.w);
      Bl[nb+ 8][bkr]=f2bf(f2.x); Bl[nb+ 9][bkr]=f2bf(f2.y);
      Bl[nb+10][bkr]=f2bf(f2.z); Bl[nb+11][bkr]=f2bf(f2.w);
      Bl[nb+12][bkr]=f2bf(f3.x); Bl[nb+13][bkr]=f2bf(f3.y);
      Bl[nb+14][bkr]=f2bf(f3.z); Bl[nb+15][bkr]=f2bf(f3.w);
    }
    __syncthreads();

    FragBF fa;
    // A frag 16x32: lane holds row laneN; K chunked per lane half
    fa.q[0] = *(const uint4*)&Al[mw + laneN][hi * 8];
    fa.q[1] = *(const uint4*)&Al[mw + laneN][16 + hi * 8];
#pragma unroll
    for (int t = 0; t < 4; ++t) {
      FragBF fb;
      // B frag 32x16: lane holds col; K chunked per lane half (same as A)
      fb.q[0] = *(const uint4*)&Bl[nw + t * 16 + laneN][hi * 8];
      fb.q[1] = *(const uint4*)&Bl[nw + t * 16 + laneN][16 + hi * 8];
      c[t].v = WMMA_BF16(fa.v, fb.v, c[t].v);
    }
  }

#pragma unroll
  for (int t = 0; t < 4; ++t) {
    const int gn = n0 + nw + t * 16 + laneN;
    const float bv = bias[gn];
#pragma unroll
    for (int r = 0; r < 8; ++r) {
      const int gm = m0 + mw + (hi ? r + 8 : r);
      const float v = c[t].f[r] + bv;
      if (OUT_F32) ((float*)outp)[(size_t)gm * N + gn] = v;
      else         ((u16*)outp)[(size_t)gm * N + gn]   = f2bf(v);
    }
  }
}

// ---------------------------------------------------------------------------
// RoPE + repack: qraw/kraw/vraw are [B*S, H] bf16 (token-major).
// Writes Qh,Kh as [B,NH,S,HD] bf16 (rope applied) and Vt as [B,NH,HD,S] bf16.
// ---------------------------------------------------------------------------
__global__ __launch_bounds__(256)
void rope_pack(const u16* __restrict__ qraw, const u16* __restrict__ kraw,
               const u16* __restrict__ vraw,
               u16* __restrict__ Qh, u16* __restrict__ Kh, u16* __restrict__ Vt)
{
  const int tid = blockIdx.x * blockDim.x + threadIdx.x;
  const int i = tid & 31;
  const int s = (tid >> 5) & 2047;
  const int h = (tid >> 16) & 15;
  const int b = tid >> 20;

  // inv_freq = 10000^(-(2i)/64) = 10000^(-i/32)
  const float inv_freq = __powf(10000.0f, -(float)i * (1.0f / 32.0f));
  const float ang = (float)s * inv_freq;
  const float c = cosf(ang), sn = sinf(ang);

  const size_t src = ((size_t)(b * 2048 + s)) * 1024 + h * 64 + i;
  const size_t dst = (((size_t)(b * 16 + h)) * 2048 + s) * 64 + i;

  const float q1 = bf2f(qraw[src]), q2 = bf2f(qraw[src + 32]);
  Qh[dst]      = f2bf(q1 * c - q2 * sn);
  Qh[dst + 32] = f2bf(q2 * c + q1 * sn);
  const float k1 = bf2f(kraw[src]), k2 = bf2f(kraw[src + 32]);
  Kh[dst]      = f2bf(k1 * c - k2 * sn);
  Kh[dst + 32] = f2bf(k2 * c + k1 * sn);

  const size_t vdst = (((size_t)(b * 16 + h)) * 64 + i) * 2048 + s;
  Vt[vdst]             = vraw[src];
  Vt[vdst + 32 * 2048] = vraw[src + 32];
}

// ---------------------------------------------------------------------------
// Flash attention, causal + pad mask, online softmax, TDM double buffering.
// Grid: (S/64, B*NH). Block: 128 threads = 4 waves; wave w owns q rows
// [q0+16w, q0+16w+16). Key loop in blocks of 32 up to end of the q block.
// ---------------------------------------------------------------------------
__global__ __launch_bounds__(128)
void flash_attn(const u16* __restrict__ Q, const u16* __restrict__ K,
                const u16* __restrict__ Vt, const float* __restrict__ amask,
                float* __restrict__ ctx)
{
  __shared__ u16 Kl[2][32][72];   // [buf][key][dim], 144B rows (16B aligned)
  __shared__ u16 Vl[2][64][40];   // [buf][dim][key], 80B rows
  __shared__ u16 Pl[4][16][40];   // per-wave probs transpose scratch

  const int tid   = threadIdx.x;
  const int lane  = tid & 31;
  const int wid   = tid >> 5;
  const int laneN = lane & 15;
  const int hi    = lane >> 4;
  const int bh = blockIdx.y;           // b*16 + h
  const int b  = bh >> 4, h = bh & 15;
  const int q0 = blockIdx.x * 64;

  // Q fragments (16 rows x 64 dims -> two 16x32 A-frags), direct from global
  FragBF fq0, fq1;
  {
    const u16* Qrow = Q + (((size_t)bh * 2048) + q0 + wid * 16 + laneN) * 64;
    fq0.q[0] = *(const uint4*)(Qrow + hi * 8);
    fq0.q[1] = *(const uint4*)(Qrow + 16 + hi * 8);
    fq1.q[0] = *(const uint4*)(Qrow + 32 + hi * 8);
    fq1.q[1] = *(const uint4*)(Qrow + 48 + hi * 8);
  }

  FragF acc[4];
  float mrow[8], lrow[8];
#pragma unroll
  for (int d = 0; d < 4; ++d)
#pragma unroll
    for (int r = 0; r < 8; ++r) acc[d].f[r] = 0.f;
#pragma unroll
  for (int r = 0; r < 8; ++r) { mrow[r] = -1e30f; lrow[r] = 0.f; }

  const int nkb = (q0 >> 5) + 2;       // keys [0, q0+64) cover all unmasked

  // --- tile staging (TDM if available, cooperative copy otherwise) ---------
#if HAVE_TDM
  #define STAGE_TILES(bufn, kbase_)                                           \
    do { if (wid == 0) {                                                      \
      tdm_load_2d((u32)(size_t)&Kl[bufn][0][0],                               \
                  K + (((size_t)bh * 2048) + (kbase_)) * 64,                  \
                  64u, 2048u, 64u, 32u, 64u, /*pad*/ 4u, 3u);                 \
      tdm_load_2d((u32)(size_t)&Vl[bufn][0][0],                               \
                  Vt + ((size_t)bh * 64) * 2048 + (kbase_),                   \
                  2048u, 64u, 32u, 64u, 2048u, /*pad*/ 3u, 3u);               \
    } } while (0)
  #define STAGE_WAIT() do { if (wid == 0) tdm_wait0(); } while (0)
#else
  #define STAGE_TILES(bufn, kbase_)                                           \
    do {                                                                      \
      { const int key = tid >> 2, seg = tid & 3;                              \
        const u16* src = K + (((size_t)bh * 2048) + (kbase_) + key) * 64      \
                           + seg * 16;                                        \
        *(uint4*)&Kl[bufn][key][seg * 16]     = *(const uint4*)src;           \
        *(uint4*)&Kl[bufn][key][seg * 16 + 8] = *(const uint4*)(src + 8); }   \
      { const int dim = tid >> 1, seg = tid & 1;                              \
        const u16* src = Vt + (((size_t)bh * 64) + dim) * 2048 + (kbase_)     \
                            + seg * 16;                                       \
        *(uint4*)&Vl[bufn][dim][seg * 16]     = *(const uint4*)src;           \
        *(uint4*)&Vl[bufn][dim][seg * 16 + 8] = *(const uint4*)(src + 8); }   \
    } while (0)
  #define STAGE_WAIT() do { } while (0)
#endif

  STAGE_TILES(0, 0);
  STAGE_WAIT();
  __syncthreads();

  for (int kb = 0; kb < nkb; ++kb) {
    const int kbase = kb * 32;
    const int cur = kb & 1;
    if (kb + 1 < nkb) STAGE_TILES(cur ^ 1, kbase + 32);

    // scores: Q(16x64) * K^T(64x32) -> two 16x16 f32 tiles
    FragF s[2];
#pragma unroll
    for (int t = 0; t < 2; ++t) {
      const int col = t * 16 + laneN;   // key within block
      FragBF fb0, fb1;
      fb0.q[0] = *(const uint4*)&Kl[cur][col][hi * 8];
      fb0.q[1] = *(const uint4*)&Kl[cur][col][16 + hi * 8];
      fb1.q[0] = *(const uint4*)&Kl[cur][col][32 + hi * 8];
      fb1.q[1] = *(const uint4*)&Kl[cur][col][48 + hi * 8];
#pragma unroll
      for (int r = 0; r < 8; ++r) s[t].f[r] = 0.f;
      s[t].v = WMMA_BF16(fq0.v, fb0.v, s[t].v);
      s[t].v = WMMA_BF16(fq1.v, fb1.v, s[t].v);
      const int key = kbase + col;
      const float padv = (1.0f - amask[b * 2048 + key]) * (-1e4f);
#pragma unroll
      for (int r = 0; r < 8; ++r) {
        const int row = q0 + wid * 16 + (hi ? r + 8 : r);
        s[t].f[r] = s[t].f[r] * 0.125f + ((key > row) ? -1e10f : 0.f) + padv;
      }
    }

    // online softmax (row stats live across each 16-lane half)
    float p0[8], p1[8], alpha[8];
#pragma unroll
    for (int r = 0; r < 8; ++r) {
      float mb = fmaxf(s[0].f[r], s[1].f[r]);
      mb = fmaxf(mb, __shfl_xor(mb, 1));
      mb = fmaxf(mb, __shfl_xor(mb, 2));
      mb = fmaxf(mb, __shfl_xor(mb, 4));
      mb = fmaxf(mb, __shfl_xor(mb, 8));
      const float mnew = fmaxf(mrow[r], mb);
      alpha[r] = __expf(mrow[r] - mnew);
      p0[r] = __expf(s[0].f[r] - mnew);
      p1[r] = __expf(s[1].f[r] - mnew);
      float ps = p0[r] + p1[r];
      ps += __shfl_xor(ps, 1);
      ps += __shfl_xor(ps, 2);
      ps += __shfl_xor(ps, 4);
      ps += __shfl_xor(ps, 8);
      lrow[r] = lrow[r] * alpha[r] + ps;
      mrow[r] = mnew;
    }
#pragma unroll
    for (int d = 0; d < 4; ++d)
#pragma unroll
      for (int r = 0; r < 8; ++r) acc[d].f[r] *= alpha[r];

    // transpose probs (C-layout -> A-layout) through per-wave LDS scratch
#pragma unroll
    for (int r = 0; r < 8; ++r) {
      const int prow = hi ? r + 8 : r;
      Pl[wid][prow][laneN]      = f2bf(p0[r]);
      Pl[wid][prow][16 + laneN] = f2bf(p1[r]);
    }
    __syncthreads();

    FragBF fp;
    fp.q[0] = *(const uint4*)&Pl[wid][laneN][hi * 8];
    fp.q[1] = *(const uint4*)&Pl[wid][laneN][16 + hi * 8];
#pragma unroll
    for (int d = 0; d < 4; ++d) {
      const int dim = d * 16 + laneN;
      FragBF fv;
      fv.q[0] = *(const uint4*)&Vl[cur][dim][hi * 8];
      fv.q[1] = *(const uint4*)&Vl[cur][dim][16 + hi * 8];
      acc[d].v = WMMA_BF16(fp.v, fv.v, acc[d].v);
    }

    STAGE_WAIT();        // next tiles resident before anyone reads them
    __syncthreads();     // also re-protects Pl / current buffers
  }

  // normalize and write ctx (f32, token-major, head cols h*64..h*64+63)
#pragma unroll
  for (int r = 0; r < 8; ++r) {
    const int row = q0 + wid * 16 + (hi ? r + 8 : r);
    const float inv = 1.0f / lrow[r];
#pragma unroll
    for (int d = 0; d < 4; ++d) {
      const int col = h * 64 + d * 16 + laneN;
      ctx[((size_t)b * 2048 + row) * 1024 + col] = acc[d].f[r] * inv;
    }
  }
}

// ---------------------------------------------------------------------------
extern "C" void kernel_launch(void* const* d_in, const int* in_sizes, int n_in,
                              void* d_out, int out_size, void* d_ws, size_t ws_size,
                              hipStream_t stream)
{
  (void)in_sizes; (void)n_in; (void)out_size; (void)ws_size;
  const float* x  = (const float*)d_in[0];
  const float* am = (const float*)d_in[1];
  const float* Wq = (const float*)d_in[2];
  const float* bq = (const float*)d_in[3];
  const float* Wk = (const float*)d_in[4];
  const float* bk = (const float*)d_in[5];
  const float* Wv = (const float*)d_in[6];
  const float* bv = (const float*)d_in[7];
  const float* Wo = (const float*)d_in[8];
  const float* bo = (const float*)d_in[9];
  float* out = (float*)d_out;

  const int B = 2, S = 2048, H = 1024, NH = 16;
  const int M = B * S;

  char* ws = (char*)d_ws;
  const size_t sz_bf = (size_t)M * H * sizeof(u16);   // 8 MB each
  u16*   qraw = (u16*)(ws);
  u16*   kraw = (u16*)(ws + 1 * sz_bf);
  u16*   vraw = (u16*)(ws + 2 * sz_bf);
  u16*   Qh   = (u16*)(ws + 3 * sz_bf);
  u16*   Kh   = (u16*)(ws + 4 * sz_bf);
  u16*   Vt   = (u16*)(ws + 5 * sz_bf);
  float* ctx  = (float*)(ws + 6 * sz_bf);             // 16 MB (total 64 MB)

  dim3 gg(H / 128, M / 64);
  gemm64_bias<false><<<gg, 256, 0, stream>>>(x, Wq, bq, qraw, M, H, H);
  gemm64_bias<false><<<gg, 256, 0, stream>>>(x, Wk, bk, kraw, M, H, H);
  gemm64_bias<false><<<gg, 256, 0, stream>>>(x, Wv, bv, vraw, M, H, H);
  rope_pack<<<(B * NH * S * 32) / 256, 256, 0, stream>>>(qraw, kraw, vraw, Qh, Kh, Vt);
  flash_attn<<<dim3(S / 64, B * NH), 128, 0, stream>>>(Qh, Kh, Vt, am, ctx);
  gemm64_bias<true><<<gg, 256, 0, stream>>>(ctx, Wo, bo, out, M, H, H);
}
